// Encoder_GAT_18726057411198
// MI455X (gfx1250) — matrix operand order repaired and checked by reading, not compiled, not run on hardware
//
#include <hip/hip_runtime.h>
#include <hip/hip_fp16.h>

// ---------------------------------------------------------------------------
// GAT encoder for MI455X (gfx1250, wave32, WMMA).
//  - all GEMMs via v_wmma_f32_16x16x32_f16 (f16 in, f32 accumulate)
//  - readout GEMM fused: one pass over 400MB graph_neigh computes
//    graph_neigh@emb, graph_neigh@emb_a and row sums simultaneously
//  - branch-free fragment loads AND stores; compile-time GEMM shapes so
//    all epilogue stores use immediate offsets
// ---------------------------------------------------------------------------

#define N_NODES 10000
#define N_PAD   10016   // padded K for transposed f16 emb (zero tail)
#define M_PAD32 10016   // rows covered by 32-row GEMM tiles (313*32)
#define M_PAD64 10048   // rows covered by 64-row readout tiles (157*64)
#define N_EDGES 160000
#define EN_TOT  (N_EDGES + N_NODES)
#define F_IN    512
#define F_OUT   256
#define SLOPE   0.2f
#define EPS_N   1e-12f

typedef __attribute__((ext_vector_type(16))) _Float16 v16h;
typedef __attribute__((ext_vector_type(8)))  _Float16 v8h;
typedef __attribute__((ext_vector_type(8)))  float    v8f;

// ---------------- order-preserving float<->uint (for atomic max) -----------
static __device__ __forceinline__ unsigned fenc(float x) {
  unsigned u = __float_as_uint(x);
  return (u & 0x80000000u) ? ~u : (u | 0x80000000u);
}
static __device__ __forceinline__ float fdec(unsigned u) {
  return (u & 0x80000000u) ? __uint_as_float(u & 0x7fffffffu)
                           : __uint_as_float(~u);
}

// ---------------- WMMA fragment builders (contiguous k order k = 16h + s, same for A and B)
// A (16x32, 16-bit): lane&15 = M row; lanes 0-15 hold K=0..15, lanes 16-31 hold K=16..31.
// Rows are CLAMPED (not guarded): out-of-range rows feed output rows that
// are never consumed, so unconditional loads vectorize to global_load_b128.
static __device__ __forceinline__ v16h a_frag_f32(const float* __restrict__ A,
                                                  int lda, int m0, int k0,
                                                  int M, int lane) {
  v16h f;
  const int r = min(m0 + (lane & 15), M - 1);
  const int h = (lane >> 4) & 1;
  const float* row = A + (size_t)r * lda + k0;
#pragma unroll
  for (int i = 0; i < 16; ++i) {
    const int kk = (h << 4) + i;
    f[i] = (_Float16)row[kk];
  }
  return f;
}

#define LDS_STR 40  // f16 row stride: 80B -> 16B aligned chunks, bank-clean
// A fragment from f16 LDS tile [rows][LDS_STR]; two aligned 16B chunks.
static __device__ __forceinline__ v16h a_frag_lds(const _Float16* __restrict__ As,
                                                  int rbase, int lane) {
  const int h = (lane >> 4) & 1;
  const _Float16* row = As + (rbase + (lane & 15)) * LDS_STR;
  const v8h lo = *reinterpret_cast<const v8h*>(row + (h << 4));        // K 0..7 / 16..23
  const v8h hi = *reinterpret_cast<const v8h*>(row + (h << 4) + 8);    // K 8..15 / 24..31
  v16h f;
#pragma unroll
  for (int i = 0; i < 8; ++i) { f[i] = lo[i]; f[i + 8] = hi[i]; }
  return f;
}

// B (32x16, 16-bit) from "B transposed" f32 storage: value(k,n) = B[n*ldb+k].
// lane&15 = column N; lanes 0-15 hold K=0..15, lanes 16-31 hold K=16..31.
static __device__ __forceinline__ v16h b_frag_bt_f32(const float* __restrict__ B,
                                                     int ldb, int k0, int n0,
                                                     int lane) {
  v16h f;
  const int n = n0 + (lane & 15);
  const int h = (lane >> 4) & 1;
  const float* row = B + (size_t)n * ldb + k0 + (h << 4);
#pragma unroll
  for (int i = 0; i < 16; ++i) f[i] = (_Float16)row[i];
  return f;
}

// B from transposed f16 storage BT[n*ldk + k] (ldk = N_PAD, zero-padded):
// 16 contiguous f16 per lane -> two aligned 16B vector loads, no guards.
static __device__ __forceinline__ v16h b_frag_hT(const _Float16* __restrict__ BT,
                                                 int ldk, int k0, int n0,
                                                 int lane) {
  const int n = n0 + (lane & 15);
  const int h = (lane >> 4) & 1;
  const _Float16* p = BT + (size_t)n * ldk + k0 + (h << 4);
  const v8h lo = *reinterpret_cast<const v8h*>(p);
  const v8h hi = *reinterpret_cast<const v8h*>(p + 8);
  v16h f;
#pragma unroll
  for (int i = 0; i < 8; ++i) { f[i] = lo[i]; f[i + 8] = hi[i]; }
  return f;
}

static __device__ __forceinline__ v8f wmma_f16(v16h a, v16h b, v8f c) {
  return __builtin_amdgcn_wmma_f32_16x16x32_f16(false, a, false, b,
                                                (short)0, c, false, false);
}

// ---------------- generic C[Mpad,NC] = A[M,KK] @ B[NC,KK]^T ----------------
// Shapes are compile-time: epilogue stores get immediate offsets, k-loop
// addressing strength-reduces against constant lda. C is padded ws memory,
// so the 8 stores are unconditional.
template <int NC, int KK>
__global__ void __launch_bounds__(256)
k_gemm_bt(const float* __restrict__ A, const float* __restrict__ B,
          float* __restrict__ C, int M) {
  const int lane = threadIdx.x & 31;
  const int wave = threadIdx.x >> 5;                 // 8 waves: 2 (M) x 4 (N)
  const int m0 = blockIdx.y * 32 + (wave >> 2) * 16;
  const int n0 = blockIdx.x * 64 + (wave & 3) * 16;
  v8f acc = {};
  for (int k0 = 0; k0 < KK; k0 += 32) {
    v16h a = a_frag_f32(A, KK, m0, k0, M, lane);
    v16h b = b_frag_bt_f32(B, KK, k0, n0, lane);
    acc = wmma_f16(a, b, acc);
  }
  const int col   = n0 + (lane & 15);
  const int rbase = m0 + ((lane >> 4) << 3);
  float* p = C + (size_t)rbase * NC + col;
#pragma unroll
  for (int j = 0; j < 8; ++j) p[j * NC] = acc[j];
}

// ---------------- fused readout GEMM -------------------------------------
// vs1 = gn@emb, vs2 = gn@emb_a, rowsum = gn@1 ; one streaming pass over gn.
// vs1/vs2/rowsum padded to M_PAD64 rows -> unconditional epilogue stores.
#define RD_ROWS 64
__global__ void __launch_bounds__(512)
k_readout_gemm(const float* __restrict__ Agn,
               const _Float16* __restrict__ B1T, const _Float16* __restrict__ B2T,
               float* __restrict__ vs1, float* __restrict__ vs2,
               float* __restrict__ rowsum) {
  __shared__ _Float16 As[RD_ROWS * LDS_STR];
  const int tid  = threadIdx.x;
  const int lane = tid & 31;
  const int wave = tid >> 5;                 // 0..15 -> column tile n0=wave*16
  const int row0 = blockIdx.x * RD_ROWS;
  const int n0   = wave * 16;
  v8f acc1[4] = {};
  v8f acc2[4] = {};
  float rs[4] = {0.f, 0.f, 0.f, 0.f};

  for (int k0 = 0; k0 < N_NODES; k0 += 32) {
    // cooperative A tile load (64x32), branch-free: clamp addr, select zero
#pragma unroll
    for (int j = 0; j < 4; ++j) {
      const int r  = row0 + wave + 16 * j;
      const int rc = min(r, N_NODES - 1);
      const int k  = k0 + lane;
      const int kc = min(k, N_NODES - 1);
      float v = Agn[(size_t)rc * N_NODES + kc];
      v = (k < N_NODES) ? v : 0.f;           // exact zero for rowsum + no NaN
      rs[j] += v;
      As[(wave + 16 * j) * LDS_STR + lane] = (_Float16)v;
      __builtin_prefetch(&Agn[(size_t)rc * N_NODES + min(k + 32, N_NODES - 1)],
                         0, 3);
    }
    __syncthreads();
    const v16h b1 = b_frag_hT(B1T, N_PAD, k0, n0, lane);
    const v16h b2 = b_frag_hT(B2T, N_PAD, k0, n0, lane);
#pragma unroll
    for (int t = 0; t < 4; ++t) {
      const v16h a = a_frag_lds(As, t * 16, lane);
      acc1[t] = wmma_f16(a, b1, acc1[t]);
      acc2[t] = wmma_f16(a, b2, acc2[t]);
    }
    __syncthreads();
  }
  // row sums: wave w owns rows {w, w+16, w+32, w+48} of this block (padded)
#pragma unroll
  for (int j = 0; j < 4; ++j) {
    float v = rs[j];
    for (int o = 16; o; o >>= 1) v += __shfl_xor(v, o, 32);
    if (lane == 0) rowsum[row0 + wave + 16 * j] = v;
  }
  const int col = n0 + (lane & 15);
#pragma unroll
  for (int t = 0; t < 4; ++t) {
    const int rbase = row0 + t * 16 + ((lane >> 4) << 3);
    float* p1 = vs1 + (size_t)rbase * F_OUT + col;
    float* p2 = vs2 + (size_t)rbase * F_OUT + col;
#pragma unroll
    for (int j = 0; j < 8; ++j) {
      p1[j * F_OUT] = acc1[t][j];
      p2[j * F_OUT] = acc2[t][j];
    }
  }
}

// ---------------- per-node attention dots: a_s = h.att_src, a_d = h.att_dst
__global__ void k_attdot(const float* __restrict__ h,
                         const float* __restrict__ avs,
                         const float* __restrict__ avd,
                         float* __restrict__ a_s, float* __restrict__ a_d,
                         int F) {
  const int gw   = (int)((blockIdx.x * (size_t)blockDim.x + threadIdx.x) >> 5);
  const int lane = threadIdx.x & 31;
  if (gw >= N_NODES) return;
  const float* row = h + (size_t)gw * F;
  float s1 = 0.f, s2 = 0.f;
  for (int f = lane; f < F; f += 32) {
    const float v = row[f];
    s1 += v * avs[f];
    s2 += v * avd[f];
  }
  for (int o = 16; o; o >>= 1) {
    s1 += __shfl_xor(s1, o, 32);
    s2 += __shfl_xor(s2, o, 32);
  }
  if (lane == 0) { a_s[gw] = s1; a_d[gw] = s2; }
}

// ---------------- edge passes ----------------------------------------------
__global__ void k_edge1(const int* __restrict__ src, const int* __restrict__ dst,
                        const float* __restrict__ a_s, const float* __restrict__ a_d,
                        float* __restrict__ ebuf, unsigned* __restrict__ menc) {
  const int e = blockIdx.x * blockDim.x + threadIdx.x;
  if (e >= EN_TOT) return;
  int s, d;
  if (e < N_EDGES) { s = src[e]; d = dst[e]; } else { s = d = e - N_EDGES; }
  float v = a_s[s] + a_d[d];
  v = (v > 0.f) ? v : SLOPE * v;
  ebuf[e] = v;
  atomicMax(&menc[d], fenc(v));
}

__global__ void k_edge2(const int* __restrict__ dst, float* __restrict__ ebuf,
                        const unsigned* __restrict__ menc,
                        float* __restrict__ ssum) {
  const int e = blockIdx.x * blockDim.x + threadIdx.x;
  if (e >= EN_TOT) return;
  const int d = (e < N_EDGES) ? dst[e] : (e - N_EDGES);
  const float p = expf(ebuf[e] - fdec(menc[d]));
  ebuf[e] = p;
  unsafeAtomicAdd(&ssum[d], p);
}

__global__ void k_edge3(const int* __restrict__ src, const int* __restrict__ dst,
                        const float* __restrict__ ebuf, const float* __restrict__ ssum,
                        const float* __restrict__ h, float* __restrict__ out, int F) {
  const int lanes = F >> 2;
  const long gid = (long)blockIdx.x * blockDim.x + threadIdx.x;
  const int e = (int)(gid / lanes);
  if (e >= EN_TOT) return;
  const int f = ((int)(gid % lanes)) << 2;
  int s, d;
  if (e < N_EDGES) { s = src[e]; d = dst[e]; } else { s = d = e - N_EDGES; }
  const float coef = ebuf[e] / ssum[d];
  const float4 hv = *reinterpret_cast<const float4*>(h + (size_t)s * F + f);
  float* o = out + (size_t)d * F + f;
  unsafeAtomicAdd(&o[0], coef * hv.x);
  unsafeAtomicAdd(&o[1], coef * hv.y);
  unsafeAtomicAdd(&o[2], coef * hv.z);
  unsafeAtomicAdd(&o[3], coef * hv.w);
}

// ---------------- small utility kernels ------------------------------------
__global__ void k_setf(float* __restrict__ p, float v, long n) {
  const long i = (long)blockIdx.x * blockDim.x + threadIdx.x;
  if (i < n) p[i] = v;
}
__global__ void k_setu(unsigned* __restrict__ p, unsigned v, long n) {
  const long i = (long)blockIdx.x * blockDim.x + threadIdx.x;
  if (i < n) p[i] = v;
}

// relu + f16 + transpose: yT[f*N_PAD + node] = relu(x[node*F_OUT + f])
__global__ void k_relucvtT(const float* __restrict__ x,
                           _Float16* __restrict__ yT) {
  const int node = blockIdx.x * blockDim.x + threadIdx.x;
  const int f    = blockIdx.y;
  if (node >= N_NODES) return;
  yT[(size_t)f * N_PAD + node] =
      (_Float16)fmaxf(x[(size_t)node * F_OUT + f], 0.f);
}

// g = sigmoid(L2-normalize(vs/rowsum)) ; in place over vs
__global__ void __launch_bounds__(256)
k_readout_fin(float* __restrict__ vs, const float* __restrict__ rowsum) {
  __shared__ float red[9];
  const int row  = blockIdx.x;
  const int tid  = threadIdx.x;
  const int lane = tid & 31;
  const float v = vs[(size_t)row * F_OUT + tid] / rowsum[row];
  float p = v * v;
  for (int o = 16; o; o >>= 1) p += __shfl_xor(p, o, 32);
  if (lane == 0) red[tid >> 5] = p;
  __syncthreads();
  if (tid == 0) {
    float t = 0.f;
    for (int i = 0; i < 8; ++i) t += red[i];
    red[8] = fmaxf(sqrtf(t), EPS_N);
  }
  __syncthreads();
  const float g = v / red[8];
  vs[(size_t)row * F_OUT + tid] = 1.f / (1.f + expf(-g));
}

// ret[n]   = {relu(z).U + bd,  relu(za).U + bd}
// ret_a[n] = {relu(za).Ua + bd, relu(z).Ua + bd}
__global__ void k_bilin(const float* __restrict__ z, const float* __restrict__ za,
                        const float* __restrict__ U, const float* __restrict__ Ua,
                        const float* __restrict__ bd,
                        float* __restrict__ ret, float* __restrict__ ret_a) {
  const int gw   = (int)((blockIdx.x * (size_t)blockDim.x + threadIdx.x) >> 5);
  const int lane = threadIdx.x & 31;
  if (gw >= N_NODES) return;
  const size_t base = (size_t)gw * F_OUT;
  float s0 = 0.f, s1 = 0.f, s2 = 0.f, s3 = 0.f;
  for (int f = lane; f < F_OUT; f += 32) {
    const float e1 = fmaxf(z[base + f], 0.f);
    const float e2 = fmaxf(za[base + f], 0.f);
    const float u  = U[base + f];
    const float ua = Ua[base + f];
    s0 += e1 * u;  s1 += e2 * u;  s2 += e2 * ua;  s3 += e1 * ua;
  }
  for (int o = 16; o; o >>= 1) {
    s0 += __shfl_xor(s0, o, 32); s1 += __shfl_xor(s1, o, 32);
    s2 += __shfl_xor(s2, o, 32); s3 += __shfl_xor(s3, o, 32);
  }
  if (lane == 0) {
    const float b = bd[0];
    ret[(size_t)gw * 2 + 0]   = s0 + b;
    ret[(size_t)gw * 2 + 1]   = s1 + b;
    ret_a[(size_t)gw * 2 + 0] = s2 + b;
    ret_a[(size_t)gw * 2 + 1] = s3 + b;
  }
}

// ---------------------------------------------------------------------------
extern "C" void kernel_launch(void* const* d_in, const int* in_sizes, int n_in,
                              void* d_out, int out_size, void* d_ws, size_t ws_size,
                              hipStream_t stream) {
  (void)in_sizes; (void)n_in; (void)out_size; (void)ws_size;
  const float* feat   = (const float*)d_in[0];
  const float* feat_a = (const float*)d_in[1];
  const int*   eidx   = (const int*)d_in[2];
  const float* gneigh = (const float*)d_in[3];
  const float* W1  = (const float*)d_in[4];
  const float* as1 = (const float*)d_in[5];
  const float* ad1 = (const float*)d_in[6];
  const float* W2  = (const float*)d_in[8];
  const float* as2 = (const float*)d_in[9];
  const float* ad2 = (const float*)d_in[10];
  const float* Wd  = (const float*)d_in[12];
  const float* bd  = (const float*)d_in[13];

  const int* src = eidx;                // edge_index[0]
  const int* dst = eidx + N_EDGES;      // edge_index[1]

  float* outz  = (float*)d_out;                      // hiden_emb [N,256]
  float* outh  = outz + (size_t)N_NODES * F_OUT;     // h         [N,512]
  float* retp  = outh + (size_t)N_NODES * F_IN;      // ret       [N,2]
  float* retap = retp + (size_t)N_NODES * 2;         // ret_a     [N,2]

  float* w = (float*)d_ws;
  size_t off = 0;
  auto alloc = [&](size_t n) { float* p = w + off; off += n; return p; };
  // GEMM destinations padded to tile-grid coverage (stores unguarded)
  float* h1     = alloc((size_t)M_PAD32 * F_OUT);
  float* h2     = alloc((size_t)M_PAD32 * F_IN);
  float* h1a    = alloc((size_t)M_PAD32 * F_OUT);
  float* za     = alloc((size_t)N_NODES * F_OUT);
  float* a_s    = alloc(N_NODES);
  float* a_d    = alloc(N_NODES);
  float* ebuf   = alloc(EN_TOT);
  unsigned* menc = (unsigned*)alloc(N_NODES);
  float* ssum   = alloc(N_NODES);
  const size_t embTfloats = (size_t)F_OUT * N_PAD / 2;   // f16 stored as floats
  _Float16* embT  = (_Float16*)alloc(embTfloats);
  _Float16* embaT = (_Float16*)alloc(embTfloats);
  float* vs1    = alloc((size_t)M_PAD64 * F_OUT);
  float* vs2    = alloc((size_t)M_PAD64 * F_OUT);
  float* rowsum = alloc(M_PAD64);
  float* U  = h1;   // reuse (h1 dead after conv1)
  float* Ua = h1a;  // reuse (h1a dead after conv on feat_a)

  auto cdiv = [](long a, long b) { return (int)((a + b - 1) / b); };

  // ---- one GAT conv layer: out = softmax-agg(x@W^T) -----------------------
  auto run_conv = [&](const float* X, int K, int Nc, const float* Wt,
                      const float* avs, const float* avd,
                      float* hbuf, float* out) {
    dim3 gg(cdiv(Nc, 64), cdiv(N_NODES, 32));
    if (Nc == F_OUT)
      k_gemm_bt<F_OUT, F_IN><<<gg, 256, 0, stream>>>(X, Wt, hbuf, N_NODES);
    else
      k_gemm_bt<F_IN, F_OUT><<<gg, 256, 0, stream>>>(X, Wt, hbuf, N_NODES);
    k_attdot<<<cdiv((long)N_NODES * 32, 256), 256, 0, stream>>>(hbuf, avs, avd,
                                                                a_s, a_d, Nc);
    k_setu<<<cdiv(N_NODES, 256), 256, 0, stream>>>(menc, 0u, N_NODES);
    k_setf<<<cdiv(N_NODES, 256), 256, 0, stream>>>(ssum, 0.f, N_NODES);
    k_setf<<<cdiv((long)N_NODES * Nc, 256), 256, 0, stream>>>(out, 0.f,
                                                              (long)N_NODES * Nc);
    k_edge1<<<cdiv(EN_TOT, 256), 256, 0, stream>>>(src, dst, a_s, a_d, ebuf, menc);
    k_edge2<<<cdiv(EN_TOT, 256), 256, 0, stream>>>(dst, ebuf, menc, ssum);
    k_edge3<<<cdiv((long)EN_TOT * (Nc / 4), 256), 256, 0, stream>>>(
        src, dst, ebuf, ssum, hbuf, out, Nc);
  };

  run_conv(feat,   F_IN,  F_OUT, W1, as1, ad1, h1,  outz);  // z = hiden_emb
  run_conv(outz,   F_OUT, F_IN,  W2, as2, ad2, h2,  outh);  // h
  run_conv(feat_a, F_IN,  F_OUT, W1, as1, ad1, h1a, za);    // z_a

  // ---- readout: g = sigmoid(normalize(gn@relu(z) / rowsum)) ---------------
  // zero padded transposed emb buffers (zero K-tail => guard-free B frags)
  k_setf<<<cdiv((long)embTfloats, 256), 256, 0, stream>>>((float*)embT, 0.f,
                                                          (long)embTfloats);
  k_setf<<<cdiv((long)embTfloats, 256), 256, 0, stream>>>((float*)embaT, 0.f,
                                                          (long)embTfloats);
  dim3 gt(cdiv(N_NODES, 256), F_OUT);
  k_relucvtT<<<gt, 256, 0, stream>>>(outz, embT);
  k_relucvtT<<<gt, 256, 0, stream>>>(za, embaT);
  k_readout_gemm<<<cdiv(N_NODES, RD_ROWS), 512, 0, stream>>>(gneigh, embT, embaT,
                                                             vs1, vs2, rowsum);
  k_readout_fin<<<N_NODES, 256, 0, stream>>>(vs1, rowsum);  // vs1 -> g
  k_readout_fin<<<N_NODES, 256, 0, stream>>>(vs2, rowsum);  // vs2 -> g_a

  // ---- bilinear: U = g @ Wd^T ; row-wise dots with relu(z)/relu(z_a) ------
  dim3 gu(cdiv(F_OUT, 64), cdiv(N_NODES, 32));
  k_gemm_bt<F_OUT, F_OUT><<<gu, 256, 0, stream>>>(vs1, Wd, U,  N_NODES);
  k_gemm_bt<F_OUT, F_OUT><<<gu, 256, 0, stream>>>(vs2, Wd, Ua, N_NODES);
  k_bilin<<<cdiv((long)N_NODES * 32, 256), 256, 0, stream>>>(outz, za, U, Ua, bd,
                                                             retp, retap);
}
